// CrossAttention_44521630990470
// MI455X (gfx1250) — matrix operand order repaired and checked by reading, not compiled, not run on hardware
//
#include <hip/hip_runtime.h>

// ---------------------------------------------------------------------------
// CrossAttention for MI455X (gfx1250).
// B=2, N=2048, Nq=1024, C=1024, H=16, D=64, SCALE=1/8.
// All matmuls: v_wmma_f32_16x16x32_bf16 (bf16 operands, fp32 accum).
// Shared operands staged in LDS via async-to-LDS (ASYNCcnt) double buffering.
// ---------------------------------------------------------------------------

typedef __bf16 bf16_t;
typedef __attribute__((ext_vector_type(16))) __bf16 v16bf;
typedef __attribute__((ext_vector_type(8)))  __bf16 v8bf;
typedef __attribute__((ext_vector_type(8)))  float  v8f;
typedef __attribute__((ext_vector_type(4)))  int    v4i;

#define IDX16 0,1,2,3,4,5,6,7,8,9,10,11,12,13,14,15
#define ATT_SCALE 0.125f

// ---- async global->LDS copy (CDNA5), with safe fallback --------------------
#if __has_builtin(__builtin_amdgcn_global_load_async_to_lds_b128)
#define HAVE_ASYNC 1
typedef __attribute__((address_space(1))) v4i g_v4i;  // prints as __device__
typedef __attribute__((address_space(3))) v4i l_v4i;  // LDS
static __device__ __forceinline__ void cp16(void* lds, const void* g) {
  // flat LDS pointer low 32 bits == LDS byte offset (aperture rule)
  __builtin_amdgcn_global_load_async_to_lds_b128(
      (g_v4i*)(size_t)g, (l_v4i*)(unsigned)(size_t)lds, 0, 0);
}
#if __has_builtin(__builtin_amdgcn_s_wait_asynccnt)
#define ASYNC_WAIT(n) __builtin_amdgcn_s_wait_asynccnt(n)
#else
#define ASYNC_WAIT(n) asm volatile("s_wait_asynccnt %0" ::"i"(n))
#endif
#else
#define HAVE_ASYNC 0
static __device__ __forceinline__ void cp16(void* lds, const void* g) {
  *(v8bf*)lds = *(const v8bf*)g;  // sync fallback through VGPRs
}
#define ASYNC_WAIT(n) do {} while (0)
#endif

// ---- small helpers ---------------------------------------------------------
static __device__ __forceinline__ v8bf cvt8(v8f f) {
  v8bf r;
#pragma unroll
  for (int i = 0; i < 8; ++i) r[i] = (__bf16)f[i];
  return r;
}
static __device__ __forceinline__ v16bf make16(v8bf lo, v8bf hi) {
  return __builtin_shufflevector(lo, hi, IDX16);
}
static __device__ __forceinline__ v8f wmma_bf16(v16bf a, v16bf b, v8f c) {
  return __builtin_amdgcn_wmma_f32_16x16x32_bf16(false, a, false, b,
                                                 (short)0, c, false, false);
}
static __device__ __forceinline__ float redmax16(float v) {
  v = fmaxf(v, __shfl_xor(v, 1, 32));
  v = fmaxf(v, __shfl_xor(v, 2, 32));
  v = fmaxf(v, __shfl_xor(v, 4, 32));
  v = fmaxf(v, __shfl_xor(v, 8, 32));
  return v;
}
static __device__ __forceinline__ float redsum16(float v) {
  v += __shfl_xor(v, 1, 32);
  v += __shfl_xor(v, 2, 32);
  v += __shfl_xor(v, 4, 32);
  v += __shfl_xor(v, 8, 32);
  return v;
}

// ---------------------------------------------------------------------------
// fp32 -> bf16 pre-conversion (memory-bound, one pass)
// ---------------------------------------------------------------------------
__global__ __launch_bounds__(256) void cvt_bf16_kernel(
    const float* __restrict__ src, bf16_t* __restrict__ dst, int n) {
  const int i = (blockIdx.x * 256 + threadIdx.x) * 8;
  if (i < n) *(v8bf*)(dst + i) = cvt8(*(const v8f*)(src + i));
}

// ---------------------------------------------------------------------------
// Unified bf16 GEMM:  out[m, c] = sum_k A[arow(m), k] * W[c, k]  (K = 1024)
//   mode 0: Q   (remap rows: first 1024 tokens/batch), bf16 out [2048,1024]
//   mode 1: K   plain, bf16 out [4096,1024]
//   mode 2: V   bf16 out transposed per head [B,H,D,N]
//   mode 3: proj, fp32 out + bias
// Block: 128 thr (4 waves). Tile 128 rows x 64 cols; wave = 32 rows.
// W tile [64 cols x 64 k] double-buffered in LDS via async copies.
// ---------------------------------------------------------------------------
__global__ __launch_bounds__(128) void gemm_kernel(
    const bf16_t* __restrict__ A, const bf16_t* __restrict__ W,
    void* __restrict__ outp, const float* __restrict__ bias, int mode) {
  __shared__ alignas(16) bf16_t Wt[2][64][64];  // 16 KB

  const int tid = threadIdx.x;
  const int lane = tid & 31, wv = tid >> 5;
  const int half = lane >> 4, l15 = lane & 15;
  const int colbase = blockIdx.y * 64;

  // this wave's two 16-row A groups
  const bf16_t* ap[2];
#pragma unroll
  for (int rg = 0; rg < 2; ++rg) {
    const int m = blockIdx.x * 128 + wv * 32 + rg * 16 + l15;
    const int ar = (mode == 0) ? ((m >> 10) * 2048 + (m & 1023)) : m;
    ap[rg] = A + (size_t)ar * 1024;
  }

  // cooperative W staging: thread -> 64B of row (tid>>1)
  const int srow = tid >> 1;
  const int scol = (tid & 1) * 32;  // elements
  const bf16_t* wsrc = W + (size_t)(colbase + srow) * 1024 + scol;
  auto stageW = [&](int buf, int k0) {
#pragma unroll
    for (int j = 0; j < 4; ++j)
      cp16(&Wt[buf][srow][scol + j * 8], wsrc + k0 + j * 8);
  };

  // A fragments (A-layout 16x32), register double buffer
  v16bf afr[2][2];
  auto loadA = [&](int k0) {
#pragma unroll
    for (int rg = 0; rg < 2; ++rg)
#pragma unroll
      for (int ks = 0; ks < 2; ++ks) {
        v8bf lo = *(const v8bf*)(ap[rg] + k0 + ks * 32 + half * 16);
        v8bf hi = *(const v8bf*)(ap[rg] + k0 + ks * 32 + half * 16 + 8);
        afr[rg][ks] = make16(lo, hi);
      }
  };

  const v8f zero = {0.f, 0.f, 0.f, 0.f, 0.f, 0.f, 0.f, 0.f};
  v8f acc[2][4];
#pragma unroll
  for (int rg = 0; rg < 2; ++rg)
#pragma unroll
    for (int ct = 0; ct < 4; ++ct) acc[rg][ct] = zero;

  float bv[4];
#pragma unroll
  for (int ct = 0; ct < 4; ++ct) bv[ct] = bias[colbase + ct * 16 + l15];

  stageW(0, 0);
  loadA(0);
  for (int kb = 0; kb < 16; ++kb) {
    const int k0 = kb * 64;
    if (kb < 15) { stageW((kb + 1) & 1, k0 + 64); ASYNC_WAIT(4); }
    else         { ASYNC_WAIT(0); }
    __syncthreads();  // W stage kb visible to all waves

    v16bf a0[2][2];
#pragma unroll
    for (int rg = 0; rg < 2; ++rg)
#pragma unroll
      for (int ks = 0; ks < 2; ++ks) a0[rg][ks] = afr[rg][ks];
    if (kb < 15) loadA(k0 + 64);  // prefetch next A while WMMAs run

#pragma unroll
    for (int ks = 0; ks < 2; ++ks)
#pragma unroll
      for (int ct = 0; ct < 4; ++ct) {
        // B-layout (32x16): lane n -> col n&15, K = (n>>4)*16 + [0,16)
        v16bf bfr =
            *(const v16bf*)&Wt[kb & 1][ct * 16 + l15][ks * 32 + half * 16];
#pragma unroll
        for (int rg = 0; rg < 2; ++rg)
          acc[rg][ct] = wmma_bf16(a0[rg][ks], bfr, acc[rg][ct]);
      }
    __syncthreads();  // all reads of buffer kb&1 done before restage
  }

#pragma unroll
  for (int rg = 0; rg < 2; ++rg)
#pragma unroll
    for (int ct = 0; ct < 4; ++ct)
#pragma unroll
      for (int g = 0; g < 8; ++g) {
        const int r = blockIdx.x * 128 + wv * 32 + rg * 16 + g + half * 8;
        const int c = colbase + ct * 16 + l15;
        const float v = acc[rg][ct][g];
        if (mode == 2) {  // V transposed: [B,H,D,N]
          const int bb = r >> 11, n = r & 2047, hh = c >> 6, d = c & 63;
          ((bf16_t*)outp)[(size_t)((bb * 16 + hh) * 64 + d) * 2048 + n] =
              (__bf16)v;
        } else if (mode == 3) {
          ((float*)outp)[(size_t)r * 1024 + c] = v + bv[ct];
        } else {
          ((bf16_t*)outp)[(size_t)r * 1024 + c] = (__bf16)v;
        }
      }
}

// ---------------------------------------------------------------------------
// Flash attention. Grid (B*H=32, Nq/64=16), 128 thr (4 waves), wave = 16 q.
// K/V 64-kv tiles double-buffered in LDS via async copies; online softmax.
// ---------------------------------------------------------------------------
__global__ __launch_bounds__(128) void attn_kernel(
    const bf16_t* __restrict__ Q, const bf16_t* __restrict__ K,
    const bf16_t* __restrict__ Vt, bf16_t* __restrict__ O) {
  __shared__ alignas(16) bf16_t Kt[2][64][64];   // 16 KB
  __shared__ alignas(16) bf16_t Vts[2][64][64];  // 16 KB
  __shared__ alignas(16) bf16_t Pst[4][16][64];  // 8 KB per-wave P staging

  const int bh = blockIdx.x;
  const int b = bh >> 4, h = bh & 15;
  const int qt = blockIdx.y;
  const int tid = threadIdx.x;
  const int lane = tid & 31, wv = tid >> 5;
  const int half = lane >> 4, l15 = lane & 15;

  // cooperative K/V staging: thread -> 64B of row (tid>>1)
  const int srow = tid >> 1;
  const int scol = (tid & 1) * 32;
  const bf16_t* ksrc = K + (size_t)(b * 2048 + srow) * 1024 + h * 64 + scol;
  const bf16_t* vsrc = Vt + (size_t)((b * 16 + h) * 64 + srow) * 2048 + scol;
  auto stageKV = [&](int buf, int t) {
#pragma unroll
    for (int j = 0; j < 4; ++j)
      cp16(&Kt[buf][srow][scol + j * 8], ksrc + (size_t)t * 64 * 1024 + j * 8);
#pragma unroll
    for (int j = 0; j < 4; ++j)
      cp16(&Vts[buf][srow][scol + j * 8], vsrc + t * 64 + j * 8);
  };

  // Q fragments (A-layout), K-dim = head dim 64
  v16bf qa[2];
  {
    const bf16_t* qb =
        Q + (size_t)(b * 1024 + qt * 64 + wv * 16 + l15) * 1024 + h * 64;
#pragma unroll
    for (int ks = 0; ks < 2; ++ks) {
      v8bf lo = *(const v8bf*)(qb + ks * 32 + half * 16);
      v8bf hi = *(const v8bf*)(qb + ks * 32 + half * 16 + 8);
      qa[ks] = make16(lo, hi);
    }
  }

  float m[8], l[8];
  const v8f zero = {0.f, 0.f, 0.f, 0.f, 0.f, 0.f, 0.f, 0.f};
  v8f acc[4] = {zero, zero, zero, zero};
#pragma unroll
  for (int g = 0; g < 8; ++g) { m[g] = -1.0e30f; l[g] = 0.0f; }

  stageKV(0, 0);
  for (int t = 0; t < 32; ++t) {
    if (t < 31) { stageKV((t + 1) & 1, t + 1); ASYNC_WAIT(8); }
    else        { ASYNC_WAIT(0); }
    __syncthreads();  // tile t staged

    // ---- S = Q @ K^T (16 x 64), fp32 ----
    v8f s[4];
#pragma unroll
    for (int ct = 0; ct < 4; ++ct) {
      v8f c = zero;
#pragma unroll
      for (int ks = 0; ks < 2; ++ks) {
        v16bf kb =
            *(const v16bf*)&Kt[t & 1][ct * 16 + l15][ks * 32 + half * 16];
        c = wmma_bf16(qa[ks], kb, c);
      }
      s[ct] = c;
    }

    // ---- online softmax (row stats shared per 16-lane half) ----
    float alpha[8];
#pragma unroll
    for (int g = 0; g < 8; ++g) {
      float vx = fmaxf(fmaxf(s[0][g], s[1][g]), fmaxf(s[2][g], s[3][g]));
      vx = redmax16(vx) * ATT_SCALE;
      const float mn = fmaxf(m[g], vx);
      alpha[g] = __expf(m[g] - mn);
      m[g] = mn;
      float rs = 0.0f;
#pragma unroll
      for (int ct = 0; ct < 4; ++ct) {
        const float p = __expf(s[ct][g] * ATT_SCALE - mn);
        s[ct][g] = p;
        rs += p;
      }
      rs = redsum16(rs);
      l[g] = l[g] * alpha[g] + rs;
#pragma unroll
      for (int ct = 0; ct < 4; ++ct) acc[ct][g] *= alpha[g];
    }

    // ---- P: C-layout fp32 -> A-layout bf16 via per-wave LDS tile ----
#pragma unroll
    for (int g = 0; g < 8; ++g)
#pragma unroll
      for (int ct = 0; ct < 4; ++ct)
        Pst[wv][g + half * 8][ct * 16 + l15] = (__bf16)s[ct][g];
    __syncthreads();  // order P writes vs reads (and pace the waves)

    // ---- acc += P @ V_tile ----
#pragma unroll
    for (int ks = 0; ks < 2; ++ks) {
      v8bf lo = *(const v8bf*)&Pst[wv][l15][ks * 32 + half * 16];
      v8bf hi = *(const v8bf*)&Pst[wv][l15][ks * 32 + half * 16 + 8];
      v16bf pa = make16(lo, hi);
#pragma unroll
      for (int ct = 0; ct < 4; ++ct) {
        v16bf vb =
            *(const v16bf*)&Vts[t & 1][ct * 16 + l15][ks * 32 + half * 16];
        acc[ct] = wmma_bf16(pa, vb, acc[ct]);
      }
    }
    __syncthreads();  // all reads of buffer t&1 done before restage
  }

  // ---- normalize and store bf16 [2048,1024] ----
#pragma unroll
  for (int g = 0; g < 8; ++g) {
    const float inv = 1.0f / l[g];
    const int r = b * 1024 + qt * 64 + wv * 16 + g + half * 8;
#pragma unroll
    for (int ct = 0; ct < 4; ++ct)
      O[(size_t)r * 1024 + h * 64 + ct * 16 + l15] =
          (__bf16)(acc[ct][g] * inv);
  }
}

// ---------------------------------------------------------------------------
extern "C" void kernel_launch(void* const* d_in, const int* in_sizes, int n_in,
                              void* d_out, int out_size, void* d_ws,
                              size_t ws_size, hipStream_t stream) {
  const float* x      = (const float*)d_in[0];
  const float* wq     = (const float*)d_in[1];
  const float* wk     = (const float*)d_in[2];
  const float* wvv    = (const float*)d_in[3];
  const float* w_proj = (const float*)d_in[4];
  const float* b_proj = (const float*)d_in[5];

  // Workspace (bf16): xb 8MB | w*b 4x2MB | Q 4MB | K 8MB | Vt 8MB | AO 4MB
  bf16_t* xb  = (bf16_t*)d_ws;
  bf16_t* wqb = xb  + (size_t)4096 * 1024;
  bf16_t* wkb = wqb + (size_t)1024 * 1024;
  bf16_t* wvb = wkb + (size_t)1024 * 1024;
  bf16_t* wpb = wvb + (size_t)1024 * 1024;
  bf16_t* Q   = wpb + (size_t)1024 * 1024;
  bf16_t* K   = Q   + (size_t)2048 * 1024;
  bf16_t* Vt  = K   + (size_t)4096 * 1024;
  bf16_t* AO  = Vt  + (size_t)4096 * 1024;

  const int NX = 4096 * 1024, NW = 1024 * 1024;
  cvt_bf16_kernel<<<NX / 2048, 256, 0, stream>>>(x, xb, NX);
  cvt_bf16_kernel<<<NW / 2048, 256, 0, stream>>>(wq, wqb, NW);
  cvt_bf16_kernel<<<NW / 2048, 256, 0, stream>>>(wk, wkb, NW);
  cvt_bf16_kernel<<<NW / 2048, 256, 0, stream>>>(wvv, wvb, NW);
  cvt_bf16_kernel<<<NW / 2048, 256, 0, stream>>>(w_proj, wpb, NW);

  const dim3 blk(128);
  gemm_kernel<<<dim3(16, 16), blk, 0, stream>>>(xb, wqb, Q,  b_proj, 0);
  gemm_kernel<<<dim3(32, 16), blk, 0, stream>>>(xb, wkb, K,  b_proj, 1);
  gemm_kernel<<<dim3(32, 16), blk, 0, stream>>>(xb, wvb, Vt, b_proj, 2);
  attn_kernel<<<dim3(32, 16), blk, 0, stream>>>(Q, K, Vt, AO);
  gemm_kernel<<<dim3(16, 16), blk, 0, stream>>>(AO, wpb, (float*)d_out,
                                                b_proj, 3);
}
